// TopkSparseAutoEncoder2Child_v2_7456063225990
// MI455X (gfx1250) — compile-verified, matrix operand order checked by reading, not run on hardware
//
#include <hip/hip_runtime.h>
#include <hip/hip_bf16.h>

#define BROWS 4096
#define DIM   768
#define SDIM  16384
#define EMA_C 0.01f

typedef __attribute__((ext_vector_type(16))) __bf16 v16bf;
typedef __attribute__((ext_vector_type(8)))  float  v8f;

union BFrag { uint4 q[2]; v16bf v; };

__device__ __forceinline__ unsigned short f2bf(float f) {
  unsigned int u = __float_as_uint(f);
  return (unsigned short)((u + 0x7FFFu + ((u >> 16) & 1u)) >> 16);
}

// ---------------- Kernel 0: convert enc_w f32 -> bf16 ----------------
__global__ void cvt_bf16_kernel(const float* __restrict__ src,
                                unsigned short* __restrict__ dst, int n) {
  int i = blockIdx.x * blockDim.x + threadIdx.x;
  if (i < n) dst[i] = f2bf(src[i]);
}

// ---------------- Kernel 1: fused encoder GEMM (WMMA bf16) + top-3 ----------------
// Block = 256 threads (8 waves), handles 32 rows of x against all 16384 columns.
// Each wave owns a 16-col tile stream; every B fragment feeds TWO WMMAs (M blocking),
// and the B fragment for step k+32 is loaded before the WMMA consuming step k.
__global__ __launch_bounds__(256)
void enc_topk_kernel(const float* __restrict__ x,
                     const unsigned short* __restrict__ wb,   // enc_w bf16 [S][D]
                     const float* __restrict__ enc_b,
                     float* __restrict__ topv, int* __restrict__ topi,
                     unsigned int* __restrict__ hit)
{
  __shared__ __align__(16) unsigned char smem[49152];
  unsigned short* sA = (unsigned short*)smem;            // 32 x 768 bf16 = 49152 B

  const int m0   = blockIdx.x * 32;
  const int tid  = threadIdx.x;
  const int lane = tid & 31;
  const int w    = tid >> 5;
  const int nl   = lane & 15;          // M for A-frag / N for B-frag / N for C
  const int sel  = (lane >> 4) & 1;    // K-half selector (ISA 16-bit frag layout)

  // Stage A tile (x rows m0..m0+31) into LDS as bf16
  for (int i = tid; i < 32 * DIM; i += 256) {
    int r = i / DIM, c = i - r * DIM;
    sA[i] = f2bf(x[(size_t)(m0 + r) * DIM + c]);
  }
  __syncthreads();

  // per-lane running top-3: 2 M-tiles x 8 row-slots (row = half*16 + slot + 8*sel)
  float tv0[2][8], tv1[2][8], tv2[2][8];
  int   ti0[2][8], ti1[2][8], ti2[2][8];
#pragma unroll
  for (int h = 0; h < 2; ++h)
#pragma unroll
    for (int r = 0; r < 8; ++r) {
      tv0[h][r] = tv1[h][r] = tv2[h][r] = -__builtin_inff();
      ti0[h][r] = ti1[h][r] = ti2[h][r] = 0;
    }

  const unsigned short* abase0 = sA + (size_t)nl * DIM + sel * 8;         // rows 0..15
  const unsigned short* abase1 = sA + (size_t)(nl + 16) * DIM + sel * 8;  // rows 16..31

  for (int nt = w; nt < SDIM / 16; nt += 8) {
    const int scol = nt * 16 + nl;
    const unsigned short* bbase = wb + (size_t)scol * DIM + sel * 8;

    v8f acc0 = {0.f, 0.f, 0.f, 0.f, 0.f, 0.f, 0.f, 0.f};
    v8f acc1 = {0.f, 0.f, 0.f, 0.f, 0.f, 0.f, 0.f, 0.f};

    // prime B double buffer
    BFrag bcur;
    bcur.q[0] = *(const uint4*)(bbase);
    bcur.q[1] = *(const uint4*)(bbase + 16);

#pragma unroll 4
    for (int kk = 0; kk < DIM; kk += 32) {
      // prefetch B for next K-step before consuming current one
      const int kn = (kk + 32 < DIM) ? (kk + 32) : 0;
      BFrag bnxt;
      bnxt.q[0] = *(const uint4*)(bbase + kn);
      bnxt.q[1] = *(const uint4*)(bbase + kn + 16);

      BFrag a0, a1;
      a0.q[0] = *(const uint4*)(abase0 + kk);
      a0.q[1] = *(const uint4*)(abase0 + kk + 16);
      a1.q[0] = *(const uint4*)(abase1 + kk);
      a1.q[1] = *(const uint4*)(abase1 + kk + 16);

      acc0 = __builtin_amdgcn_wmma_f32_16x16x32_bf16(
          false, a0.v, false, bcur.v, (short)0, acc0, false, false);
      acc1 = __builtin_amdgcn_wmma_f32_16x16x32_bf16(
          false, a1.v, false, bcur.v, (short)0, acc1, false, false);

      bcur = bnxt;
    }

    const float bias = enc_b[scol];
#pragma unroll
    for (int r = 0; r < 8; ++r) {
      float va = acc0[r] + bias;
      if (va > tv2[0][r]) {
        if (va > tv0[0][r]) {
          tv2[0][r] = tv1[0][r]; ti2[0][r] = ti1[0][r];
          tv1[0][r] = tv0[0][r]; ti1[0][r] = ti0[0][r];
          tv0[0][r] = va;        ti0[0][r] = scol;
        } else if (va > tv1[0][r]) {
          tv2[0][r] = tv1[0][r]; ti2[0][r] = ti1[0][r];
          tv1[0][r] = va;        ti1[0][r] = scol;
        } else {
          tv2[0][r] = va;        ti2[0][r] = scol;
        }
      }
      float vb = acc1[r] + bias;
      if (vb > tv2[1][r]) {
        if (vb > tv0[1][r]) {
          tv2[1][r] = tv1[1][r]; ti2[1][r] = ti1[1][r];
          tv1[1][r] = tv0[1][r]; ti1[1][r] = ti0[1][r];
          tv0[1][r] = vb;        ti0[1][r] = scol;
        } else if (vb > tv1[1][r]) {
          tv2[1][r] = tv1[1][r]; ti2[1][r] = ti1[1][r];
          tv1[1][r] = vb;        ti1[1][r] = scol;
        } else {
          tv2[1][r] = vb;        ti2[1][r] = scol;
        }
      }
    }
  }

  // sA dead; reuse LDS for candidate merge, one 16-row half at a time
  float* candV = (float*)smem;             // 16 rows * 384 cand * 4B = 24576
  int*   candI = (int*)(smem + 24576);     // 24576
#pragma unroll
  for (int half = 0; half < 2; ++half) {
    __syncthreads();
#pragma unroll
    for (int r = 0; r < 8; ++r) {
      int m    = r + 8 * sel;
      int base = m * 384 + w * 48 + nl * 3;
      candV[base + 0] = tv0[half][r]; candV[base + 1] = tv1[half][r]; candV[base + 2] = tv2[half][r];
      candI[base + 0] = ti0[half][r]; candI[base + 1] = ti1[half][r]; candI[base + 2] = ti2[half][r];
    }
    __syncthreads();
    if (tid < 16) {
      float v0 = -__builtin_inff(), v1 = v0, v2 = v0;
      int   i0 = 0, i1 = 0, i2 = 0;
      const int base = tid * 384;
      for (int c = 0; c < 384; ++c) {
        float v = candV[base + c];
        int  ix = candI[base + c];
        if (v > v2) {
          if (v > v0)      { v2 = v1; i2 = i1; v1 = v0; i1 = i0; v0 = v; i0 = ix; }
          else if (v > v1) { v2 = v1; i2 = i1; v1 = v;  i1 = ix; }
          else             { v2 = v;  i2 = ix; }
        }
      }
      const int row = m0 + half * 16 + tid;
      topv[row * 3 + 0] = v0; topv[row * 3 + 1] = v1; topv[row * 3 + 2] = v2;
      topi[row * 3 + 0] = i0; topi[row * 3 + 1] = i1; topi[row * 3 + 2] = i2;
      hit[i0] = 1u; hit[i1] = 1u; hit[i2] = 1u;   // racy 1-stores: fine
    }
  }
}

// ---------------- Kernel 2: child dots + masks + sparse reconstruction ----------------
__global__ __launch_bounds__(256)
void child_recon_kernel(const float* __restrict__ x,
                        const float* __restrict__ enc1_w, const float* __restrict__ enc1_b,
                        const float* __restrict__ enc2_w, const float* __restrict__ enc2_b,
                        const float* __restrict__ dec_w,  const float* __restrict__ dec_b,
                        const float* __restrict__ dec1_w, const float* __restrict__ dec1_b,
                        const float* __restrict__ dec2_w, const float* __restrict__ dec2_b,
                        const float* __restrict__ topv, const int* __restrict__ topi,
                        unsigned int* __restrict__ live1, unsigned int* __restrict__ live2,
                        float4* __restrict__ rec, float* __restrict__ out)
{
  __shared__ float red[256];
  __shared__ float sP[3], sF1[3], sF2[3];
  __shared__ int   sI[3];
  __shared__ float sDots[6];

  const int b   = blockIdx.x;
  const int tid = threadIdx.x;

  if (tid < 3) { sI[tid] = topi[b * 3 + tid]; sP[tid] = topv[b * 3 + tid]; }
  __syncthreads();

  // 6 dot products of length 768: (enc1,enc2) x 3 top-k indices
  for (int q = 0; q < 6; ++q) {
    const int j = q >> 1;
    const float* wrow = ((q & 1) ? enc2_w : enc1_w) + (size_t)sI[j] * DIM;
    float p = 0.f;
    for (int d = tid; d < DIM; d += 256) p += x[(size_t)b * DIM + d] * wrow[d];
    red[tid] = p;
    __syncthreads();
    for (int s2 = 128; s2 > 0; s2 >>= 1) {
      if (tid < s2) red[tid] += red[tid + s2];
      __syncthreads();
    }
    if (tid == 0) sDots[q] = red[0];
    __syncthreads();
  }

  if (tid == 0) {
#pragma unroll
    for (int j = 0; j < 3; ++j) {
      float p    = sP[j];
      float pre1 = sDots[2 * j]     + enc1_b[sI[j]];
      float pre2 = sDots[2 * j + 1] + enc2_b[sI[j]];
      float m1 = (p != 0.f) ? pre1 : 0.f;
      float m2 = (p != 0.f) ? pre2 : 0.f;
      bool win = m1 > m2;
      float f1 = win ? m1 : 0.f;
      float f2 = win ? 0.f : m2;
      sF1[j] = f1; sF2[j] = f2;
      if (f1 != 0.f) live1[sI[j]] = 1u;
      if (f2 != 0.f) live2[sI[j]] = 1u;
      rec[b * 3 + j] = make_float4(__int_as_float(sI[j]), p, f1, f2);
    }
  }
  __syncthreads();

  // reconstructed[b][d] = dec_b+dec1_b+dec2_b + sum_j p*decW[:,i] + f1*dec1W[:,i] + f2*dec2W[:,i]
  for (int d = tid; d < DIM; d += 256) {
    float a = dec_b[d] + dec1_b[d] + dec2_b[d];
#pragma unroll
    for (int j = 0; j < 3; ++j) {
      size_t off = (size_t)d * SDIM + sI[j];
      a += sP[j] * dec_w[off] + sF1[j] * dec1_w[off] + sF2[j] * dec2_w[off];
    }
    out[(size_t)b * DIM + d] = a;
  }
}

// ---------------- Kernel 3: EMA ratios (one thread owns one sae index) ----------------
__global__ __launch_bounds__(256)
void ema_kernel(const float4* __restrict__ rec,
                const float* __restrict__ r1_in, const float* __restrict__ r2_in,
                float* __restrict__ r1_out, float* __restrict__ r2_out)
{
  __shared__ __align__(16) float4 chunk[2048];   // 32 KB
  const int s = blockIdx.x * 256 + threadIdx.x;
  float r1 = r1_in[s], r2 = r2_in[s];
  const int total = BROWS * 3;                   // 12288 records, in row order
  for (int base = 0; base < total; base += 2048) {
    for (int i = threadIdx.x; i < 2048; i += 256) chunk[i] = rec[base + i];
    __syncthreads();
    for (int i = 0; i < 2048; ++i) {
      float4 e = chunk[i];
      if (__float_as_int(e.x) == s) {
        float p = e.y;
        if (p != 0.f) {
          if (e.z != 0.f) r1 = (1.f - EMA_C) * r1 + EMA_C * e.z / p;
          if (e.w != 0.f) r2 = (1.f - EMA_C) * r2 + EMA_C * e.w / p;
        }
      }
    }
    __syncthreads();
  }
  r1_out[s] = r1;
  r2_out[s] = r2;
}

// ---------------- Kernel 4: live counts ----------------
__global__ __launch_bounds__(256)
void count_kernel(const unsigned int* __restrict__ hit,
                  const unsigned int* __restrict__ live1,
                  const unsigned int* __restrict__ live2,
                  float* __restrict__ outc)
{
  __shared__ unsigned int c[3];
  if (threadIdx.x < 3) c[threadIdx.x] = 0u;
  __syncthreads();
  unsigned int a = 0, b1 = 0, b2 = 0;
  for (int i = threadIdx.x; i < SDIM; i += 256) {
    a  += hit[i];
    b1 += live1[i];
    b2 += live2[i];
  }
  atomicAdd(&c[0], a); atomicAdd(&c[1], b1); atomicAdd(&c[2], b2);
  __syncthreads();
  if (threadIdx.x < 3) outc[threadIdx.x] = (float)c[threadIdx.x];
}

// ---------------- launch ----------------
extern "C" void kernel_launch(void* const* d_in, const int* in_sizes, int n_in,
                              void* d_out, int out_size, void* d_ws, size_t ws_size,
                              hipStream_t stream)
{
  const float* x      = (const float*)d_in[0];
  const float* enc_w  = (const float*)d_in[1];
  const float* enc_b  = (const float*)d_in[2];
  const float* dec_w  = (const float*)d_in[3];
  const float* dec_b  = (const float*)d_in[4];
  const float* enc1_w = (const float*)d_in[5];
  const float* enc1_b = (const float*)d_in[6];
  const float* dec1_w = (const float*)d_in[7];
  const float* dec1_b = (const float*)d_in[8];
  const float* enc2_w = (const float*)d_in[9];
  const float* enc2_b = (const float*)d_in[10];
  const float* dec2_w = (const float*)d_in[11];
  const float* dec2_b = (const float*)d_in[12];
  const float* ratio1 = (const float*)d_in[13];
  const float* ratio2 = (const float*)d_in[14];
  (void)in_sizes; (void)n_in; (void)out_size; (void)ws_size;

  char* ws = (char*)d_ws;
  size_t o = 0;
  unsigned short* wb = (unsigned short*)(ws + o); o += (size_t)SDIM * DIM * 2;  // 25.2 MB
  float* topv        = (float*)(ws + o);          o += (size_t)BROWS * 3 * 4;
  int*   topi        = (int*)(ws + o);            o += (size_t)BROWS * 3 * 4;
  unsigned int* hit  = (unsigned int*)(ws + o);   o += (size_t)SDIM * 4;
  unsigned int* liv1 = (unsigned int*)(ws + o);   o += (size_t)SDIM * 4;
  unsigned int* liv2 = (unsigned int*)(ws + o);   o += (size_t)SDIM * 4;
  float4* rec        = (float4*)(ws + o);         o += (size_t)BROWS * 3 * 16;

  float* out    = (float*)d_out;
  float* recon  = out;                                 // [4096*768]
  float* counts = out + (size_t)BROWS * DIM;           // 3 scalars
  float* r1o    = counts + 3;                          // [16384]
  float* r2o    = r1o + SDIM;                          // [16384]

  // hit/liv1/liv2 are contiguous: single async memset (graph-capturable)
  hipMemsetAsync(hit, 0, (size_t)3 * SDIM * 4, stream);

  cvt_bf16_kernel<<<(SDIM * DIM + 255) / 256, 256, 0, stream>>>(enc_w, wb, SDIM * DIM);

  enc_topk_kernel<<<BROWS / 32, 256, 0, stream>>>(x, wb, enc_b, topv, topi, hit);

  child_recon_kernel<<<BROWS, 256, 0, stream>>>(x, enc1_w, enc1_b, enc2_w, enc2_b,
                                                dec_w, dec_b, dec1_w, dec1_b,
                                                dec2_w, dec2_b,
                                                topv, topi, liv1, liv2, rec, recon);

  ema_kernel<<<SDIM / 256, 256, 0, stream>>>(rec, ratio1, ratio2, r1o, r2o);

  count_kernel<<<1, 256, 0, stream>>>(hit, liv1, liv2, counts);
}
